// MetalMoEExperts_90580860272681
// MI455X (gfx1250) — matrix operand order repaired and checked by reading, not compiled, not run on hardware
//
#include <hip/hip_runtime.h>
#include <hip/hip_bf16.h>
#include <hip/hip_fp16.h>

typedef __attribute__((ext_vector_type(16))) _Float16 v16h;
typedef __attribute__((ext_vector_type(8)))  float    v8f;

#define E_    8
#define I_    2816
#define H_    2048
#define GS_   32
#define P_    2048
#define TOPK_ 2
#define NPAIR (P_*TOPK_)          // 4096
#define CAP_  NPAIR               // worst-case bucket capacity per expert

#define BUCKET_OFF 4096
#define ACT_OFF    (1u<<20)

#define MT1 4                     // gemm1: 4 x 16 = 64 pair rows per wave
#define MT2 8                     // gemm2: 8 x 16 = 128 pair rows per wave

// ---------------- routing: bucket pair ids by expert ----------------
__global__ void moe_route(const int* __restrict__ eidx,
                          int* __restrict__ counts,
                          int* __restrict__ buckets) {
    int p = blockIdx.x * blockDim.x + threadIdx.x;
    if (p >= NPAIR) return;
    int e = eidx[p];
    int pos = atomicAdd(&counts[e], 1);
    buckets[e * CAP_ + pos] = p;
}

// ---------------- GEMM1: gate_up + SiLU, fused ----------------
// wave = 64(pairs) x 16(cols of I); one dequantized gate/up B tile feeds 4 A subtiles
__global__ void __launch_bounds__(256)
moe_gemm1(const float* __restrict__ x,
          const int*   __restrict__ w1,
          const float* __restrict__ s1,
          const float* __restrict__ b1,
          const int*   __restrict__ counts,
          const int*   __restrict__ buckets,
          _Float16*    __restrict__ act)
{
    const int e     = blockIdx.z;
    const int cnt   = counts[e];
    const int mBase = blockIdx.y * (16 * MT1);
    if (mBase >= cnt) return;

    const int wave = threadIdx.x >> 5;
    const int lane = threadIdx.x & 31;
    const int mn   = lane & 15;          // A-row / B-col / C-col selector
    const int hs   = lane >> 4;          // half-wave select (K interleave)
    const int nBase = blockIdx.x * 128 + wave * 16;
    const int bktBase = e * CAP_;

    // A row pointers for the 4 subtiles (row clamped -> always safe to read;
    // invalid rows are masked at the store, WMMA rows are independent)
    const float* xr[MT1];
    #pragma unroll
    for (int t = 0; t < MT1; ++t) {
        int r = mBase + t * 16 + mn;
        r = (r < cnt) ? r : (cnt - 1);
        const int pid = buckets[bktBase + r];
        xr[t] = x + (long)(pid / TOPK_) * H_;
    }

    // B rows (gate / up) for this lane
    const long rG = (long)e * (2*I_) + (nBase + mn);
    const long rU = rG + I_;
    const int*   wg = w1 + rG * H_;
    const int*   wu = w1 + rU * H_;
    const float* sg = s1 + rG * (H_/GS_);
    const float* bg = b1 + rG * (H_/GS_);
    const float* su = s1 + rU * (H_/GS_);
    const float* bu = b1 + rU * (H_/GS_);

    v8f accG[MT1] = {};
    v8f accU[MT1] = {};

    for (int k0 = 0; k0 < H_; k0 += 32) {
        const int g = k0 >> 5;           // quant group == K-step
        // ---- shared B tiles: dequant int4 -> f16 ----
        const float scg = sg[g], ofg = bg[g];
        const float scu = su[g], ofu = bu[g];
        v16h bgv, buv;
        {
            const int* pw = wg + k0 + hs * 16;
            #pragma unroll
            for (int i = 0; i < 16; ++i)
                bgv[i] = (_Float16)((float)pw[i] * scg + ofg);
        }
        {
            const int* pw = wu + k0 + hs * 16;
            #pragma unroll
            for (int i = 0; i < 16; ++i)
                buv[i] = (_Float16)((float)pw[i] * scu + ofu);
        }
        if (k0 + 32 < H_) {
            __builtin_prefetch(wg + k0 + 32 + hs * 16, 0, 1);
            __builtin_prefetch(wu + k0 + 32 + hs * 16, 0, 1);
        }
        // ---- 4 A subtiles, 8 WMMAs, branch-free inner body ----
        #pragma unroll
        for (int t = 0; t < MT1; ++t) {
            v16h a;
            const float* pa = xr[t] + k0 + hs * 16;
            #pragma unroll
            for (int i = 0; i < 8; ++i) a[i]     = (_Float16)pa[i];
            #pragma unroll
            for (int i = 0; i < 8; ++i) a[8 + i] = (_Float16)pa[8 + i];
            accG[t] = __builtin_amdgcn_wmma_f32_16x16x32_f16(
                          false, a, false, bgv, (short)0, accG[t], false, false);
            accU[t] = __builtin_amdgcn_wmma_f32_16x16x32_f16(
                          false, a, false, buv, (short)0, accU[t], false, false);
        }
    }

    // ---- epilogue: activated = silu(gate) * up -> act[pair, col] (f16) ----
    const int ncol = nBase + mn;         // C-matrix: lane holds column mn
    #pragma unroll
    for (int t = 0; t < MT1; ++t) {
        if (mBase + t * 16 < cnt) {
            #pragma unroll
            for (int j = 0; j < 8; ++j) {
                const int mr = mBase + t * 16 + j + 8 * hs;   // C row for VGPR j
                if (mr < cnt) {
                    const int pid = buckets[bktBase + mr];
                    const float gv = accG[t][j];
                    const float uv = accU[t][j];
                    const float av = (gv / (1.0f + __expf(-gv))) * uv;
                    act[(long)pid * I_ + ncol] = (_Float16)av;
                }
            }
        }
    }
}

// ---------------- GEMM2: down-projection + weighted scatter-add ----------------
// wave = 128(pairs) x 16(cols of H); one dequantized B tile feeds 8 A subtiles
__global__ void __launch_bounds__(256)
moe_gemm2(const _Float16* __restrict__ act,
          const int*   __restrict__ w2,
          const float* __restrict__ s2,
          const float* __restrict__ b2,
          const float* __restrict__ ew,     // expert_weights flat (P*TOPK)
          const int*   __restrict__ counts,
          const int*   __restrict__ buckets,
          float*       __restrict__ out)
{
    const int e     = blockIdx.z;
    const int cnt   = counts[e];
    const int mBase = blockIdx.y * (16 * MT2);
    if (mBase >= cnt) return;

    const int wave = threadIdx.x >> 5;
    const int lane = threadIdx.x & 31;
    const int mn   = lane & 15;
    const int hs   = lane >> 4;
    const int nBase = blockIdx.x * 128 + wave * 16;   // column in [0, H)
    const int bktBase = e * CAP_;

    const _Float16* ar[MT2];
    #pragma unroll
    for (int t = 0; t < MT2; ++t) {
        int r = mBase + t * 16 + mn;
        r = (r < cnt) ? r : (cnt - 1);
        const int pid = buckets[bktBase + r];
        ar[t] = act + (long)pid * I_;
    }

    const long rr = (long)e * H_ + (nBase + mn);
    const int*   w  = w2 + rr * I_;
    const float* sr = s2 + rr * (I_/GS_);
    const float* br = b2 + rr * (I_/GS_);

    v8f acc[MT2] = {};

    for (int k0 = 0; k0 < I_; k0 += 32) {
        const int g = k0 >> 5;
        // ---- shared B tile: dequant int4 -> f16 ----
        const float sc = sr[g], of = br[g];
        v16h bv;
        {
            const int* pw = w + k0 + hs * 16;
            #pragma unroll
            for (int i = 0; i < 16; ++i)
                bv[i] = (_Float16)((float)pw[i] * sc + of);
        }
        if (k0 + 32 < I_) __builtin_prefetch(w + k0 + 32 + hs * 16, 0, 1);
        // ---- 8 A subtiles (f16 already, pure loads), branch-free ----
        #pragma unroll
        for (int t = 0; t < MT2; ++t) {
            v16h a;
            const _Float16* pa = ar[t] + k0 + hs * 16;
            #pragma unroll
            for (int i = 0; i < 8; ++i) a[i]     = pa[i];
            #pragma unroll
            for (int i = 0; i < 8; ++i) a[8 + i] = pa[8 + i];
            acc[t] = __builtin_amdgcn_wmma_f32_16x16x32_f16(
                         false, a, false, bv, (short)0, acc[t], false, false);
        }
    }

    const int ncol = nBase + mn;
    #pragma unroll
    for (int t = 0; t < MT2; ++t) {
        if (mBase + t * 16 < cnt) {
            #pragma unroll
            for (int j = 0; j < 8; ++j) {
                const int mr = mBase + t * 16 + j + 8 * hs;
                if (mr < cnt) {
                    const int pid   = buckets[bktBase + mr];
                    const int token = pid / TOPK_;
                    const float v   = acc[t][j] * ew[pid];
                    atomicAdd(&out[(long)token * H_ + ncol], v);
                }
            }
        }
    }
}

// ---------------- launcher ----------------
extern "C" void kernel_launch(void* const* d_in, const int* in_sizes, int n_in,
                              void* d_out, int out_size, void* d_ws, size_t ws_size,
                              hipStream_t stream) {
    const float* x    = (const float*)d_in[0];
    const float* ew   = (const float*)d_in[1];
    const int*   eidx = (const int*)d_in[2];
    // d_in[3] = top_k scalar (constant 2 in this problem)
    const int*   w1   = (const int*)d_in[4];
    const float* s1   = (const float*)d_in[5];
    const float* b1   = (const float*)d_in[6];
    const int*   w2   = (const int*)d_in[7];
    const float* s2   = (const float*)d_in[8];
    const float* b2   = (const float*)d_in[9];
    float* out = (float*)d_out;

    int*      counts  = (int*)d_ws;
    int*      buckets = (int*)((char*)d_ws + BUCKET_OFF);
    _Float16* act     = (_Float16*)((char*)d_ws + ACT_OFF);

    hipMemsetAsync(counts, 0, E_ * sizeof(int), stream);
    hipMemsetAsync(out, 0, (size_t)P_ * H_ * sizeof(float), stream);

    moe_route<<<(NPAIR + 255) / 256, 256, 0, stream>>>(eidx, counts, buckets);

    dim3 g1(I_ / 128, CAP_ / (16 * MT1), E_);   // (22, 64, 8)
    moe_gemm1<<<g1, 256, 0, stream>>>(x, w1, s1, b1, counts, buckets, act);

    dim3 g2(H_ / 128, CAP_ / (16 * MT2), E_);   // (16, 32, 8)
    moe_gemm2<<<g2, 256, 0, stream>>>(act, w2, s2, b2, ew, counts, buckets, out);
}
